// im2latex_45148696216215
// MI455X (gfx1250) — matrix operand order repaired and checked by reading, not compiled, not run on hardware
//
#include <hip/hip_runtime.h>
#include <math.h>

// ---------------------------------------------------------------------------
// im2latex full pipeline for gfx1250 (MI455X).
// All matrix math on v_wmma_f32_16x16x32_bf16 (fp32 accum, bf16 operands).
// 128-thread blocks (4 waves, wave32): GEMM tile 16x64 (shared A),
// conv tile 64x16 (shared im2col B). K-chunk = 64 -> 2 WMMAs per barrier
// pair, uniform 32-wide tail for K % 64 == 32 shapes.
// Attention operands (enc_w1, enc) stored bf16: halves the dominant HBM
// traffic (2x 39MB/step over 128 steps). Transcendentals on v_tanh/v_exp.
// ---------------------------------------------------------------------------

typedef __attribute__((ext_vector_type(16))) __bf16 v16bf;
typedef __attribute__((ext_vector_type(8)))  float  v8f;

#define BB   32
#define TT   128
#define VV   512
#define EE   32
#define ENC_ 256
#define DEC_ 512
#define LL   1200   // 10*120 encoder positions

#if defined(__has_builtin)
#if __has_builtin(__builtin_amdgcn_tanhf)
#define HAVE_TANH_BUILTIN 1
#endif
#endif

__device__ __forceinline__ float fast_tanh(float x) {
#ifdef HAVE_TANH_BUILTIN
  return __builtin_amdgcn_tanhf(x);     // v_tanh_f32 (CDNA5 TRANS op)
#else
  const float e = __expf(2.f * x);      // v_exp_f32; inf-safe form
  return 1.f - 2.f / (e + 1.f);
#endif
}

__device__ __forceinline__ float sigm(float x) { return 1.f / (1.f + __expf(-x)); }

__device__ __forceinline__ unsigned pack_bf16(float f0, float f1) {
  union { __bf16 h[2]; unsigned u; } p;
  p.h[0] = (__bf16)f0;
  p.h[1] = (__bf16)f1;
  return p.u;
}
__device__ __forceinline__ float bf_lo(unsigned u) {
  union { float f; unsigned x; } c; c.x = u << 16; return c.f;
}
__device__ __forceinline__ float bf_hi(unsigned u) {
  union { float f; unsigned x; } c; c.x = u & 0xffff0000u; return c.f;
}

// ---------------------------------------------------------------------------
// One wave: acc += A(16x32 bf16) * B(32x16 bf16); tiles in LDS as packed-bf16
// u32 pairs, row-major [row][32 u32], koff (u32 pairs) selects 32-wide chunk.
// Per CDNA5 ISA 7.12.2 16-bit layouts (pair index = k/2):
//   A: lane l -> m=l&15, half=l>>4; vgpr v<4: pair v+4*half ; v>=4: pair v+4+4*half
//   B: lane l -> n=l&15, half=l>>4; vgpr v<4: pair v+4*half ; v>=4: pair v+4+4*half
// ---------------------------------------------------------------------------
__device__ __forceinline__ v8f wmma_acc32(const unsigned* __restrict__ A32,
                                          const unsigned* __restrict__ B32,
                                          v8f acc, int lane, int koff) {
  const int half = lane >> 4;
  const int mr   = lane & 15;
  union { v16bf v; unsigned u[8]; } a, b;
#pragma unroll
  for (int v = 0; v < 8; ++v) {
    const int ap = (v < 4) ? (v + 4 * half) : (v + 4 + 4 * half);
    a.u[v] = A32[mr * 32 + koff + ap];
    b.u[v] = B32[mr * 32 + koff + ap];
  }
  return __builtin_amdgcn_wmma_f32_16x16x32_bf16(
      /*neg_a=*/false, a.v, /*neg_b=*/false, b.v,
      /*c_mod=*/(short)0, acc, /*reuse_a=*/false, /*reuse_b=*/false);
}

// ---------------------------------------------------------------------------
// GEMM: C[M,N] = act( X[M,K] @ W[N,K]^T (+bias) (+C) ), tile 16(M) x 64(N).
// Requires: M%16==0, N%64==0, K%32==0 (true for every call in this model).
// FLAGS: bit0 = accumulate into C, bit1 = tanh. HASB: bias present.
// BF16OUT: store result as bf16 (C is __bf16*).
// ---------------------------------------------------------------------------
template <int FLAGS, bool HASB, bool BF16OUT>
__global__ void __launch_bounds__(128) gemm_xwT_wmma(
    const float* __restrict__ X, const float* __restrict__ Wt,
    const float* __restrict__ bias, float* __restrict__ C,
    int M, int N, int K) {
  __shared__ unsigned Als32[16 * 32];   // 16 m-rows x 64 bf16 (shared by 4 waves)
  __shared__ unsigned Bls32[64 * 32];   // 64 n-rows x 64 bf16 (16 rows per wave)
  const int tid = threadIdx.x;
  const int lane = tid & 31, wv = tid >> 5;
  const int n0 = blockIdx.x << 6;
  const int m0 = blockIdx.y << 4;
  const int am = tid >> 3;              // 0..15
  const int ak = (tid & 7) << 2;        // 0,4,...,28
  const float* Arow = X + (size_t)(m0 + am) * K + ak;
  v8f acc = {};
  for (int k0 = 0; k0 < K; k0 += 64) {
    const bool two = (k0 + 32 < K);     // uniform across block
    {
      const float4 f = *(const float4*)(Arow + k0);
      *(uint2*)&Als32[am * 32 + (ak >> 1)] =
          make_uint2(pack_bf16(f.x, f.y), pack_bf16(f.z, f.w));
      if (two) {
        const float4 g = *(const float4*)(Arow + k0 + 32);
        *(uint2*)&Als32[am * 32 + 16 + (ak >> 1)] =
            make_uint2(pack_bf16(g.x, g.y), pack_bf16(g.z, g.w));
      }
    }
#pragma unroll
    for (int j = 0; j < 4; ++j) {
      const int idx = j * 128 + tid;
      const int bn = idx >> 3, bk = (idx & 7) << 2;
      const float* Brow = Wt + (size_t)(n0 + bn) * K + k0 + bk;
      const float4 f = *(const float4*)Brow;
      *(uint2*)&Bls32[bn * 32 + (bk >> 1)] =
          make_uint2(pack_bf16(f.x, f.y), pack_bf16(f.z, f.w));
      if (two) {
        const float4 g = *(const float4*)(Brow + 32);
        *(uint2*)&Bls32[bn * 32 + 16 + (bk >> 1)] =
            make_uint2(pack_bf16(g.x, g.y), pack_bf16(g.z, g.w));
      }
    }
    __syncthreads();
    const unsigned* Bw = Bls32 + wv * 16 * 32;
    acc = wmma_acc32(Als32, Bw, acc, lane, 0);
    if (two) acc = wmma_acc32(Als32, Bw, acc, lane, 16);
    __syncthreads();
  }
  const int nr = lane & 15, half = lane >> 4;
  const int n = n0 + wv * 16 + nr;
  const float bv = HASB ? bias[n] : 0.f;
#pragma unroll
  for (int v = 0; v < 8; ++v) {
    const int m = m0 + v + 8 * half;
    float r = acc[v] + bv;
    const size_t o = (size_t)m * N + n;
    if (FLAGS & 1) r += C[o];
    if (FLAGS & 2) r = fast_tanh(r);
    if (BF16OUT) ((__bf16*)C)[o] = (__bf16)r;
    else         C[o] = r;
  }
}

// ---------------------------------------------------------------------------
// Implicit-GEMM 3x3 VALID conv + bias + ReLU, NCHW / OIHW.
// Block computes 64 couts x 16 pixels; im2col gather shared by all 4 waves.
// KPAD=true only for conv1 (K=9 < 32, Cout=32 < 64): guarded staging/store.
// ---------------------------------------------------------------------------
template <bool KPAD>
__global__ void __launch_bounds__(128) conv3x3_relu_wmma(
    const float* __restrict__ in, const float* __restrict__ w,
    const float* __restrict__ bias, float* __restrict__ out,
    int N, int Cin, int IH, int IW, int Cout, int OH, int OW) {
  __shared__ unsigned Als32[64 * 32];   // 64 cout-rows x 64 bf16
  __shared__ unsigned Bls32[16 * 32];   // 16 pixel-rows x 64 bf16 (shared)
  const int K   = Cin * 9;
  const int Kc  = KPAD ? 32 : K;        // conv1: one padded chunk
  const int OHW = OH * OW;
  const int IHW = IH * IW;
  const int tid = threadIdx.x, lane = tid & 31, wv = tid >> 5;
  const int p0 = blockIdx.x << 4;
  const int c0 = blockIdx.y << 6;
  const int bn  = tid >> 3;             // 0..15 (this thread's pixel)
  const int bk0 = (tid & 7) << 2;       // 0,4,...,28
  const int p  = p0 + bn;
  const int ni = p / OHW;
  const int pr = p - ni * OHW;
  const int oh = pr / OW, ow = pr - oh * OW;
  const size_t base = (size_t)ni * Cin * IHW + (size_t)oh * IW + ow;
  v8f acc = {};
  for (int k0 = 0; k0 < Kc; k0 += 64) {
    const bool two = (k0 + 32 < Kc);    // uniform; false for conv1
#pragma unroll
    for (int j = 0; j < 4; ++j) {
      const int idx = j * 128 + tid;
      const int am = idx >> 3, ak = (idx & 7) << 2;   // am 0..63
      if (KPAD) {
        float f[4];
#pragma unroll
        for (int q = 0; q < 4; ++q) {
          const int k = k0 + ak + q;
          f[q] = (k < K && (c0 + am) < Cout) ? w[(size_t)(c0 + am) * K + k] : 0.f;
        }
        *(uint2*)&Als32[am * 32 + (ak >> 1)] =
            make_uint2(pack_bf16(f[0], f[1]), pack_bf16(f[2], f[3]));
      } else {
        const float* Wr = w + (size_t)(c0 + am) * K + k0 + ak;
        const float4 f = *(const float4*)Wr;
        *(uint2*)&Als32[am * 32 + (ak >> 1)] =
            make_uint2(pack_bf16(f.x, f.y), pack_bf16(f.z, f.w));
        if (two) {
          const float4 g = *(const float4*)(Wr + 32);
          *(uint2*)&Als32[am * 32 + 16 + (ak >> 1)] =
              make_uint2(pack_bf16(g.x, g.y), pack_bf16(g.z, g.w));
        }
      }
    }
#pragma unroll
    for (int s = 0; s < 2; ++s) {
      if (s == 1 && !two) break;
      float f[4];
      int kq  = k0 + s * 32 + bk0;
      int cin = kq / 9;
      int r   = kq - cin * 9;
#pragma unroll
      for (int q = 0; q < 4; ++q) {
        const int fh = r / 3, fw = r - fh * 3;
        f[q] = (!KPAD || kq < K)
                   ? in[base + (size_t)cin * IHW + fh * IW + fw]
                   : 0.f;
        ++kq;
        if (++r == 9) { r = 0; ++cin; }
      }
      *(uint2*)&Bls32[bn * 32 + s * 16 + (bk0 >> 1)] =
          make_uint2(pack_bf16(f[0], f[1]), pack_bf16(f[2], f[3]));
    }
    __syncthreads();
    const unsigned* Aw = Als32 + wv * 16 * 32;
    acc = wmma_acc32(Aw, Bls32, acc, lane, 0);
    if (two) acc = wmma_acc32(Aw, Bls32, acc, lane, 16);
    __syncthreads();
  }
  const int nr = lane & 15, half = lane >> 4;
  const int pp  = p0 + nr;
  const int ni2 = pp / OHW;
  const int pr2 = pp - ni2 * OHW;
  const int oh2 = pr2 / OW, ow2 = pr2 - oh2 * OW;
#pragma unroll
  for (int v = 0; v < 8; ++v) {
    const int cc = c0 + wv * 16 + v + 8 * half;
    if (!KPAD || cc < Cout) {
      const float r = acc[v] + bias[cc];
      out[(((size_t)ni2 * Cout + cc) * OH + oh2) * OW + ow2] = fmaxf(r, 0.f);
    }
  }
}

// ---------------------------------------------------------------------------
// Max pool (VALID), NCHW flattened to [NC, H, W].
// ---------------------------------------------------------------------------
__global__ void maxpool_k(const float* __restrict__ in, float* __restrict__ out,
                          int NC, int IH, int IW, int kh, int kw, int sh, int sw,
                          int OH, int OW) {
  const int total = NC * OH * OW;
  const int idx = blockIdx.x * blockDim.x + threadIdx.x;
  if (idx >= total) return;
  const int OHW = OH * OW;
  const int nc = idx / OHW;
  const int r  = idx - nc * OHW;
  const int oh = r / OW, ow = r - oh * OW;
  float m = -3.4e38f;
  for (int i = 0; i < kh; ++i)
    for (int j = 0; j < kw; ++j)
      m = fmaxf(m, in[((size_t)nc * IH + (oh * sh + i)) * IW + (ow * sw + j)]);
  out[idx] = m;
}

// x5 [B,256,10,120] -> enc [B,L,256] (fp32) + encb (bf16 copy for attention)
__global__ void nchw_to_nlc(const float* __restrict__ x, float* __restrict__ enc,
                            __bf16* __restrict__ encb) {
  const size_t total = (size_t)BB * LL * ENC_;
  const size_t idx = (size_t)blockIdx.x * blockDim.x + threadIdx.x;
  if (idx >= total) return;
  const int c = (int)(idx & (ENC_ - 1));
  const int bl = (int)(idx >> 8);
  const int l = bl % LL;
  const int b = bl / LL;
  const int h = l / 120, wq = l - h * 120;
  const float v = x[(((size_t)b * ENC_ + c) * 10 + h) * 120 + wq];
  enc[idx] = v;
  encb[idx] = (__bf16)v;
}

// mean over L: enc [B,L,256] -> mean [B,256]
__global__ void mean_enc(const float* __restrict__ enc, float* __restrict__ mn) {
  const int b = blockIdx.x, c = threadIdx.x;  // 256 threads
  float s = 0.f;
  for (int l = 0; l < LL; ++l) s += enc[((size_t)b * LL + l) * ENC_ + c];
  mn[b * ENC_ + c] = s * (1.f / (float)LL);
}

// X = [emb[formulas[:,t]], o_prev]  -> [B, E+DEC] = [32, 544]
__global__ void build_x(const float* __restrict__ emb, const int* __restrict__ formulas,
                        const float* __restrict__ o, float* __restrict__ X, int t) {
  const int idx = blockIdx.x * blockDim.x + threadIdx.x;
  if (idx >= BB * (EE + DEC_)) return;
  const int b = idx / (EE + DEC_);
  const int j = idx - b * (EE + DEC_);
  X[idx] = (j < EE) ? emb[(size_t)formulas[b * TT + t] * EE + j]
                    : o[b * DEC_ + (j - EE)];
}

// X2 = [h, ctx] -> [32, 768]
__global__ void build_x2(const float* __restrict__ h, const float* __restrict__ ctx,
                         float* __restrict__ X2) {
  const int idx = blockIdx.x * blockDim.x + threadIdx.x;
  if (idx >= BB * (DEC_ + ENC_)) return;
  const int b = idx / (DEC_ + ENC_);
  const int j = idx - b * (DEC_ + ENC_);
  X2[idx] = (j < DEC_) ? h[b * DEC_ + j] : ctx[b * ENC_ + (j - DEC_)];
}

// gates [32,2048] -> update c,h in place. split order: i,f,g,og
__global__ void lstm_elem(const float* __restrict__ gates, float* __restrict__ c,
                          float* __restrict__ h) {
  const int idx = blockIdx.x * blockDim.x + threadIdx.x;
  if (idx >= BB * DEC_) return;
  const int b = idx >> 9, j = idx & (DEC_ - 1);
  const float* g = gates + (size_t)b * 4 * DEC_;
  const float ig = g[j], fg = g[DEC_ + j], gg = g[2 * DEC_ + j], og = g[3 * DEC_ + j];
  const float cn = sigm(fg) * c[idx] + sigm(ig) * fast_tanh(gg);
  c[idx] = cn;
  h[idx] = sigm(og) * fast_tanh(cn);
}

// Fused attention per batch row (bf16 enc_w1 / enc; fp32 math):
//   e[l] = sum_c tanh(encw1[b,l,c] + hw2[b,c]) * beta[c] ; a = softmax_l(e)
//   ctx[b,c] = sum_l a[l] * enc[b,l,c]
__global__ void attn_fused(const __bf16* __restrict__ encw1, const float* __restrict__ hw2,
                           const float* __restrict__ beta, const __bf16* __restrict__ encb,
                           float* __restrict__ ctx) {
  const int b = blockIdx.x;
  const int tid = threadIdx.x;          // 256 threads = 8 waves (wave32)
  const int lane = tid & 31, wid = tid >> 5;
  __shared__ float e_s[LL];
  __shared__ float hw2_s[ENC_];
  __shared__ float beta_s[ENC_];
  __shared__ float red_s[256];
  hw2_s[tid] = hw2[b * ENC_ + tid];
  beta_s[tid] = beta[tid];
  __syncthreads();
  // Pass 1: e[l]. Each lane owns 8 contiguous channels -> one b128 load/row.
  for (int l = wid; l < LL; l += 8) {
    const uint4 pk = *(const uint4*)(encw1 + ((size_t)b * LL + l) * ENC_ + lane * 8);
    const unsigned pu[4] = {pk.x, pk.y, pk.z, pk.w};
    float p = 0.f;
#pragma unroll
    for (int j = 0; j < 4; ++j) {
      const int cc = lane * 8 + 2 * j;
      p += fast_tanh(bf_lo(pu[j]) + hw2_s[cc]) * beta_s[cc];
      p += fast_tanh(bf_hi(pu[j]) + hw2_s[cc + 1]) * beta_s[cc + 1];
    }
#pragma unroll
    for (int off = 16; off > 0; off >>= 1) p += __shfl_down(p, off, 32);
    if (lane == 0) e_s[l] = p;
  }
  __syncthreads();
  // Softmax over e_s[0..LL)
  float mx = -3.4e38f;
  for (int l = tid; l < LL; l += 256) mx = fmaxf(mx, e_s[l]);
  red_s[tid] = mx; __syncthreads();
  for (int s = 128; s > 0; s >>= 1) {
    if (tid < s) red_s[tid] = fmaxf(red_s[tid], red_s[tid + s]);
    __syncthreads();
  }
  mx = red_s[0]; __syncthreads();
  float sum = 0.f;
  for (int l = tid; l < LL; l += 256) {
    const float ex = __expf(e_s[l] - mx);
    e_s[l] = ex; sum += ex;
  }
  red_s[tid] = sum; __syncthreads();
  for (int s = 128; s > 0; s >>= 1) {
    if (tid < s) red_s[tid] += red_s[tid + s];
    __syncthreads();
  }
  const float inv = 1.f / red_s[0]; __syncthreads();
  // Pass 2: ctx. Threads 0..127 each own a bf16 channel pair (u32 loads).
  if (tid < 128) {
    const unsigned* ep = (const unsigned*)(encb + (size_t)b * LL * ENC_) + tid;
    float a0 = 0.f, a1 = 0.f;
    for (int l = 0; l < LL; ++l) {
      const unsigned u = ep[(size_t)l * (ENC_ / 2)];
      const float al = e_s[l];
      a0 += al * bf_lo(u);
      a1 += al * bf_hi(u);
    }
    ctx[b * ENC_ + 2 * tid]     = a0 * inv;
    ctx[b * ENC_ + 2 * tid + 1] = a1 * inv;
  }
}

// row softmax over V=512, write to d_out[b, t, :]
__global__ void softmax_out(const float* __restrict__ lraw, float* __restrict__ out, int t) {
  const int b = blockIdx.x, tid = threadIdx.x;  // 256 threads, 2 elems each
  __shared__ float red_s[256];
  const float* r = lraw + (size_t)b * VV;
  float mx = fmaxf(r[tid], r[tid + 256]);
  red_s[tid] = mx; __syncthreads();
  for (int s = 128; s > 0; s >>= 1) {
    if (tid < s) red_s[tid] = fmaxf(red_s[tid], red_s[tid + s]);
    __syncthreads();
  }
  mx = red_s[0]; __syncthreads();
  const float e0 = __expf(r[tid] - mx), e1 = __expf(r[tid + 256] - mx);
  red_s[tid] = e0 + e1; __syncthreads();
  for (int s = 128; s > 0; s >>= 1) {
    if (tid < s) red_s[tid] += red_s[tid + s];
    __syncthreads();
  }
  const float inv = 1.f / red_s[0];
  float* dst = out + ((size_t)b * TT + t) * VV;
  dst[tid] = e0 * inv;
  dst[tid + 256] = e1 * inv;
}

// ---------------------------------------------------------------------------
// Host-side launch sequence
// ---------------------------------------------------------------------------
static inline void gemm(const float* X, const float* W, const float* bias, float* C,
                        int M, int N, int K, int flags, hipStream_t s) {
  dim3 g(N / 64, M / 16);
  if (bias) {
    if (flags == 0)      gemm_xwT_wmma<0, true , false><<<g, 128, 0, s>>>(X, W, bias, C, M, N, K);
    else if (flags == 1) gemm_xwT_wmma<1, true , false><<<g, 128, 0, s>>>(X, W, bias, C, M, N, K);
    else                 gemm_xwT_wmma<2, true , false><<<g, 128, 0, s>>>(X, W, bias, C, M, N, K);
  } else {
    if (flags == 2)      gemm_xwT_wmma<2, false, false><<<g, 128, 0, s>>>(X, W, bias, C, M, N, K);
    else                 gemm_xwT_wmma<0, false, false><<<g, 128, 0, s>>>(X, W, bias, C, M, N, K);
  }
}

static inline void gemm_bf16out(const float* X, const float* W, __bf16* C,
                                int M, int N, int K, hipStream_t s) {
  dim3 g(N / 64, M / 16);
  gemm_xwT_wmma<0, false, true><<<g, 128, 0, s>>>(X, W, nullptr, (float*)C, M, N, K);
}

template <bool KPAD>
static inline void conv(const float* in, const float* w, const float* b, float* out,
                        int N, int Cin, int IH, int IW, int Cout, int OH, int OW,
                        hipStream_t s) {
  const int P = N * OH * OW;
  dim3 g(P / 16, (Cout + 63) / 64);
  conv3x3_relu_wmma<KPAD><<<g, 128, 0, s>>>(in, w, b, out, N, Cin, IH, IW, Cout, OH, OW);
}

extern "C" void kernel_launch(void* const* d_in, const int* in_sizes, int n_in,
                              void* d_out, int out_size, void* d_ws, size_t ws_size,
                              hipStream_t stream) {
  (void)in_sizes; (void)n_in; (void)out_size; (void)ws_size;
  const float* imgs     = (const float*)d_in[0];
  const int*   formulas = (const int*)d_in[1];
  const float* c1w = (const float*)d_in[2];  const float* c1b = (const float*)d_in[3];
  const float* c2w = (const float*)d_in[4];  const float* c2b = (const float*)d_in[5];
  const float* c3w = (const float*)d_in[6];  const float* c3b = (const float*)d_in[7];
  const float* c4w = (const float*)d_in[8];  const float* c4b = (const float*)d_in[9];
  const float* c5w = (const float*)d_in[10]; const float* c5b = (const float*)d_in[11];
  const float* emb  = (const float*)d_in[12];
  const float* wh_w = (const float*)d_in[13]; const float* wh_b = (const float*)d_in[14];
  const float* wc_w = (const float*)d_in[15]; const float* wc_b = (const float*)d_in[16];
  const float* wo_w = (const float*)d_in[17]; const float* wo_b = (const float*)d_in[18];
  const float* beta = (const float*)d_in[19];
  const float* w1   = (const float*)d_in[20];
  const float* w2   = (const float*)d_in[21];
  const float* w3   = (const float*)d_in[22];
  const float* wout = (const float*)d_in[23];
  const float* wih  = (const float*)d_in[24]; const float* whh = (const float*)d_in[25];
  const float* bih  = (const float*)d_in[26]; const float* bhh = (const float*)d_in[27];
  float* out = (float*)d_out;

  char* ws = (char*)d_ws;
  size_t off = 0;
  auto alloc = [&](size_t bytes) -> void* {
    void* p = (void*)(ws + off);
    off += (bytes + 255) & ~(size_t)255;
    return p;
  };
  float*  bufA   = (float*)alloc(129515520);           // ping (max 32*32*62*510 f32)
  float*  bufB   = (float*)alloc(50331648);            // pong (max 32*128*25*122 f32)
  float*  enc    = (float*)alloc((size_t)BB * LL * ENC_ * 4);
  __bf16* encb   = (__bf16*)alloc((size_t)BB * LL * ENC_ * 2);
  __bf16* encw1b = (__bf16*)alloc((size_t)BB * LL * ENC_ * 2);
  float*  mn     = (float*)alloc(BB * ENC_ * 4);
  float*  h      = (float*)alloc(BB * DEC_ * 4);
  float*  c      = (float*)alloc(BB * DEC_ * 4);
  float*  o      = (float*)alloc(BB * DEC_ * 4);
  float*  Xb     = (float*)alloc(BB * (EE + DEC_) * 4);
  float*  gates  = (float*)alloc(BB * 4 * DEC_ * 4);
  float*  hw2    = (float*)alloc(BB * ENC_ * 4);
  float*  ctx    = (float*)alloc(BB * ENC_ * 4);
  float*  X2     = (float*)alloc(BB * (DEC_ + ENC_) * 4);
  float*  lraw   = (float*)alloc(BB * VV * 4);

  // ---- CNN encoder ----
  conv<true >(imgs, c1w, c1b, bufA, BB, 1, 64, 512, 32, 62, 510, stream);
  {
    int NC = BB * 32, tot = NC * 31 * 255;
    maxpool_k<<<(tot + 255) / 256, 256, 0, stream>>>(bufA, bufB, NC, 62, 510, 2, 2, 2, 2, 31, 255);
  }
  conv<false>(bufB, c2w, c2b, bufA, BB, 32, 31, 255, 64, 29, 253, stream);
  {
    int NC = BB * 64, tot = NC * 29 * 126;
    maxpool_k<<<(tot + 255) / 256, 256, 0, stream>>>(bufA, bufB, NC, 29, 253, 1, 2, 1, 2, 29, 126);
  }
  conv<false>(bufB, c3w, c3b, bufA, BB, 64, 29, 126, 128, 27, 124, stream);
  conv<false>(bufA, c4w, c4b, bufB, BB, 128, 27, 124, 128, 25, 122, stream);
  {
    int NC = BB * 128, tot = NC * 12 * 122;
    maxpool_k<<<(tot + 255) / 256, 256, 0, stream>>>(bufB, bufA, NC, 25, 122, 2, 1, 2, 1, 12, 122);
  }
  conv<false>(bufA, c5w, c5b, bufB, BB, 128, 12, 122, 256, 10, 120, stream);
  {
    size_t tot = (size_t)BB * LL * ENC_;
    nchw_to_nlc<<<(unsigned)((tot + 255) / 256), 256, 0, stream>>>(bufB, enc, encb);
  }

  // ---- decoder init ----
  mean_enc<<<BB, ENC_, 0, stream>>>(enc, mn);
  gemm(mn, wh_w, wh_b, h, BB, DEC_, ENC_, /*tanh*/2, stream);
  gemm(mn, wc_w, wc_b, c, BB, DEC_, ENC_, 2, stream);
  gemm(mn, wo_w, wo_b, o, BB, DEC_, ENC_, 2, stream);
  gemm_bf16out(enc, w1, encw1b, BB * LL, ENC_, ENC_, stream);  // enc @ w1^T -> bf16

  // ---- decoder loop (teacher forcing, T=128 steps) ----
  for (int t = 0; t < TT; ++t) {
    build_x<<<(BB * (EE + DEC_) + 255) / 256, 256, 0, stream>>>(emb, formulas, o, Xb, t);
    gemm(Xb, wih, bih, gates, BB, 4 * DEC_, EE + DEC_, 0, stream);
    gemm(h, whh, bhh, gates, BB, 4 * DEC_, DEC_, /*accum*/1, stream);
    lstm_elem<<<(BB * DEC_ + 255) / 256, 256, 0, stream>>>(gates, c, h);
    gemm(h, w2, nullptr, hw2, BB, ENC_, DEC_, 0, stream);
    attn_fused<<<BB, 256, 0, stream>>>(encw1b, hw2, beta, encb, ctx);
    build_x2<<<(BB * (DEC_ + ENC_) + 255) / 256, 256, 0, stream>>>(h, ctx, X2);
    gemm(X2, w3, nullptr, o, BB, DEC_, DEC_ + ENC_, 2, stream);
    gemm(o, wout, nullptr, lraw, BB, VV, DEC_, 0, stream);
    softmax_out<<<BB, 256, 0, stream>>>(lraw, out, t);
  }
}
